// Attention_17832704213516
// MI455X (gfx1250) — matrix operand order repaired and checked by reading, not compiled, not run on hardware
//
#include <hip/hip_runtime.h>
#include <hip/hip_bf16.h>

// ---------------- problem constants ----------------
constexpr int B_   = 2;
constexpr int L_   = 2048;
constexpr int HID_ = 2048;
constexpr int H_   = 32;
constexpr int HKV_ = 8;
constexpr int D_   = 64;

// ---------------- WMMA fragment types ----------------
typedef __attribute__((ext_vector_type(16))) __bf16 v16bf;
typedef __attribute__((ext_vector_type(8)))  __bf16 v8bf;
typedef __attribute__((ext_vector_type(4)))  __bf16 v4bf;
typedef __attribute__((ext_vector_type(8)))  float  v8f;
typedef __attribute__((ext_vector_type(4)))  int    v4i;

#define WMMA_BF16(a, b, c) \
  __builtin_amdgcn_wmma_f32_16x16x32_bf16(false, (a), false, (b), (short)0, (c), false, false)

#if __has_builtin(__builtin_amdgcn_global_load_async_to_lds_b128)
#define HAVE_ASYNC_LDS 1
#else
#define HAVE_ASYNC_LDS 0
#endif

#if __has_builtin(__builtin_amdgcn_s_wait_asynccnt)
#define WAIT_ASYNC(n) __builtin_amdgcn_s_wait_asynccnt(n)
#else
#define WAIT_ASYNC(n) asm volatile("s_wait_asynccnt " #n ::: "memory")
#endif

// A-fragment (16x32, bf16): lane row = lane&15; per lane two contiguous runs of 8
// at k = 8*g and k = 16 + 8*g  (g = lane>>4).  `tile` points at [row0][k0].
// Works for global or LDS (generic) pointers.
__device__ __forceinline__ v16bf load_a(const __bf16* tile, int ld, int lane) {
  const int c = lane & 15, g = lane >> 4;
  const __bf16* p = tile + (size_t)c * ld + 8 * g;
  v8bf lo = *(const v8bf*)p;
  v8bf hi = *(const v8bf*)(p + 16);
  v16bf f;
#pragma unroll
  for (int i = 0; i < 8; ++i) { f[i] = lo[i]; f[i + 8] = hi[i]; }
  return f;
}

// B-fragment (32x16, bf16): lane col n = lane&15; per lane two contiguous runs of 8
// at k = 8*g and k = 16 + 8*g.  Storage: element (n, k) at tile[n*ld + k].
__device__ __forceinline__ v16bf load_b(const __bf16* tile, int ld, int lane) {
  const int c = lane & 15, g = lane >> 4;
  const __bf16* p = tile + (size_t)c * ld + 8 * g;
  v8bf lo = *(const v8bf*)p;
  v8bf hi = *(const v8bf*)(p + 16);
  v16bf f;
#pragma unroll
  for (int i = 0; i < 8; ++i) { f[i] = lo[i]; f[i + 8] = hi[i]; }
  return f;
}

// Stage one 32(row) x 64(k) bf16 A-chunk (4KB) into LDS, 128 threads x 2 x 16B.
// Async path uses GLOBAL_LOAD_ASYNC_TO_LDS_B128 (ASYNCcnt); fallback bounces
// through VGPRs. gsrc points at [row0][k0] with row stride ldg (elements).
__device__ __forceinline__ void stage_a_chunk(const __bf16* gsrc, int ldg,
                                              __bf16* lbuf, int tid) {
#pragma unroll
  for (int half = 0; half < 2; ++half) {
    const int o   = tid * 16 + half * 2048;  // byte offset within 4KB chunk
    const int row = o >> 7;                  // 128 bytes per row
    const int col = o & 127;
    const char* gp = (const char*)gsrc + (size_t)row * ldg * 2 + col;
    char* lp = (char*)lbuf + o;
#if HAVE_ASYNC_LDS
    __builtin_amdgcn_global_load_async_to_lds_b128((v4i*)gp, (v4i*)lp, 0, 0);
#else
    *(uint4*)lp = *(const uint4*)gp;
#endif
  }
}

// ---------------- kernel 0: fp32 -> bf16 conversion ----------------
__global__ void __launch_bounds__(256) cvt4_kernel(const float* __restrict__ src,
                                                   __bf16* __restrict__ dst, int n4) {
  int i = blockIdx.x * 256 + threadIdx.x;
  if (i < n4) {
    float4 v = ((const float4*)src)[i];
    v4bf o;
    o[0] = (__bf16)v.x; o[1] = (__bf16)v.y; o[2] = (__bf16)v.z; o[3] = (__bf16)v.w;
    ((v4bf*)dst)[i] = o;
  }
}

// ---------------- kernel 1: QKV projection + RoPE ----------------
// Block = 4 waves sharing one 32-row A tile (double-buffered async LDS pipeline).
// Each wave computes a 32(M) x 64(N) strip = one head's full D, as 2x4 WMMA
// accumulators; each B-fragment feeds two WMMAs. RoPE's d <-> d+-32 pairing is
// lane-local across tiles t and t+-2.
// strips: 0..31 -> Q head, 32..39 -> K head, 40..47 -> V head (stored transposed).
__global__ void __launch_bounds__(128) qkv_rope_kernel(
    const __bf16* __restrict__ xb,
    const __bf16* __restrict__ Wqb, const __bf16* __restrict__ Wkb,
    const __bf16* __restrict__ Wvb,
    const float* __restrict__ cosb, const float* __restrict__ sinb,
    __bf16* __restrict__ Qb, __bf16* __restrict__ Kb, __bf16* __restrict__ Vtb) {
  const int tid  = threadIdx.x;
  const int lane = tid & 31;
  const int wave = tid >> 5;
  const int c = lane & 15, g = lane >> 4;
  const int rowTile = blockIdx.x;            // 0 .. B*L/32-1
  const int strip   = blockIdx.y * 4 + wave; // 0 .. 47

  __shared__ alignas(16) __bf16 abuf[2][32 * 64];

  const __bf16* W; int head, mode;
  if (strip < 32)      { W = Wqb; head = strip;      mode = 0; }
  else if (strip < 40) { W = Wkb; head = strip - 32; mode = 1; }
  else                 { W = Wvb; head = strip - 40; mode = 2; }

  const int grow = rowTile * 32;
  const int b    = grow / L_;
  const int pos0 = grow % L_;

  const __bf16* Arow = xb + (size_t)grow * HID_;
  const __bf16* Brow = W + (size_t)head * 64 * HID_;

  v8f acc[2][4] = {};
  constexpr int NCH = HID_ / 64;  // 32 chunks of K=64

  stage_a_chunk(Arow, HID_, &abuf[0][0], tid);
  for (int j = 0; j < NCH; ++j) {
    if (j + 1 < NCH) {
      stage_a_chunk(Arow + (j + 1) * 64, HID_, &abuf[(j + 1) & 1][0], tid);
      WAIT_ASYNC(2);   // chunk j landed; chunk j+1 still in flight
    } else {
      WAIT_ASYNC(0);
    }
    __syncthreads();
    const __bf16* lb = &abuf[j & 1][0];
#pragma unroll
    for (int s = 0; s < 64; s += 32) {
      v16bf a0 = load_a(lb + s, 64, lane);            // rows 0..15
      v16bf a1 = load_a(lb + 16 * 64 + s, 64, lane);  // rows 16..31
#pragma unroll
      for (int t = 0; t < 4; ++t) {
        v16bf bf = load_b(Brow + (size_t)(t * 16) * HID_ + j * 64 + s, HID_, lane);
        acc[0][t] = WMMA_BF16(a0, bf, acc[0][t]);
        acc[1][t] = WMMA_BF16(a1, bf, acc[1][t]);
      }
    }
    __syncthreads();  // everyone done with abuf[j&1] before it is restaged
  }

#pragma unroll
  for (int m = 0; m < 2; ++m) {
    const int prow = pos0 + m * 16;
    if (mode != 2) {
      // RoPE: out[d] = x[d]*cos + rot_half(x)[d]*sin ; pairs live in tiles t, t+-2.
      float rot[4][8];
#pragma unroll
      for (int t = 0; t < 4; ++t)
#pragma unroll
        for (int r = 0; r < 8; ++r) {
          int d   = t * 16 + c;
          int pos = prow + r + 8 * g;
          float co = cosb[(size_t)pos * 64 + d];
          float si = sinb[(size_t)pos * 64 + d];
          float other = (t < 2) ? -acc[m][t + 2][r] : acc[m][t - 2][r];
          rot[t][r] = acc[m][t][r] * co + other * si;
        }
      __bf16* dst = (mode == 0) ? (Qb + ((size_t)b * H_   + head) * L_ * 64)
                                : (Kb + ((size_t)b * HKV_ + head) * L_ * 64);
#pragma unroll
      for (int t = 0; t < 4; ++t)
#pragma unroll
        for (int r = 0; r < 8; ++r)
          dst[(size_t)(prow + r + 8 * g) * 64 + t * 16 + c] = (__bf16)rot[t][r];
    } else {
      // V stored transposed [B, HKV, D, L] so P*V B-fragments read contiguously.
      __bf16* dst = Vtb + ((size_t)b * HKV_ + head) * 64 * L_;
#pragma unroll
      for (int t = 0; t < 4; ++t)
#pragma unroll
        for (int r = 0; r < 8; ++r)
          dst[(size_t)(t * 16 + c) * L_ + prow + r + 8 * g] = (__bf16)acc[m][t][r];
    }
  }
}

// ---------------- kernel 2: causal flash attention (GQA) ----------------
// One wave per (b, h, 16-query tile). Online softmax; P goes C-layout -> LDS
// (bf16) -> A-layout fragment for the P*V WMMAs.
__global__ void __launch_bounds__(128) attn_kernel(
    const __bf16* __restrict__ Qb, const __bf16* __restrict__ Kb,
    const __bf16* __restrict__ Vtb, __bf16* __restrict__ Aob) {
  const int lane = threadIdx.x & 31;
  const int wave = threadIdx.x >> 5;
  const int c = lane & 15, g = lane >> 4;

  __shared__ alignas(16) __bf16 psh[4][16 * 32];
  __bf16* pl = &psh[wave][0];

  const int QT  = L_ / 16;
  const int gid = blockIdx.x * 4 + wave;
  const int qt  = gid % QT;
  const int bh  = gid / QT;
  const int h   = bh % H_;
  const int b   = bh / H_;
  const int hkv = h / (H_ / HKV_);

  const __bf16* Qbase = Qb + (((size_t)b * H_ + h) * L_ + (size_t)qt * 16) * 64;
  const __bf16* Kbase = Kb + ((size_t)b * HKV_ + hkv) * L_ * 64;
  const __bf16* Vbase = Vtb + ((size_t)b * HKV_ + hkv) * 64 * L_;

  v16bf qf0 = load_a(Qbase, 64, lane);       // d = 0..31
  v16bf qf1 = load_a(Qbase + 32, 64, lane);  // d = 32..63

  v8f o[4] = {};
  float mrow[8], lrow[8];
#pragma unroll
  for (int r = 0; r < 8; ++r) { mrow[r] = -1e30f; lrow[r] = 0.f; }

  const float scale = 0.125f;  // 64^-0.5
  const int qpos0 = qt * 16;
  const int qend  = qpos0 + 16;

  for (int k0 = 0; k0 < qend; k0 += 32) {
    v8f s0 = {}, s1 = {};
    {
      v16bf kf0 = load_b(Kbase + (size_t)k0 * 64, 64, lane);
      v16bf kf1 = load_b(Kbase + (size_t)k0 * 64 + 32, 64, lane);
      s0 = WMMA_BF16(qf0, kf0, s0);
      s0 = WMMA_BF16(qf1, kf1, s0);
    }
    {
      v16bf kf0 = load_b(Kbase + (size_t)(k0 + 16) * 64, 64, lane);
      v16bf kf1 = load_b(Kbase + (size_t)(k0 + 16) * 64 + 32, 64, lane);
      s1 = WMMA_BF16(qf0, kf0, s1);
      s1 = WMMA_BF16(qf1, kf1, s1);
    }
#pragma unroll
    for (int r = 0; r < 8; ++r) {
      const int qpos = qpos0 + r + 8 * g;
      float x0 = (k0 + c      <= qpos) ? s0[r] * scale : -1e30f;
      float x1 = (k0 + 16 + c <= qpos) ? s1[r] * scale : -1e30f;
      float mx = fmaxf(x0, x1);
#pragma unroll
      for (int off = 1; off < 16; off <<= 1) mx = fmaxf(mx, __shfl_xor(mx, off, 32));
      float mnew  = fmaxf(mrow[r], mx);
      float alpha = __expf(mrow[r] - mnew);
      float p0 = __expf(x0 - mnew);
      float p1 = __expf(x1 - mnew);
      pl[(r + 8 * g) * 32 + c]      = (__bf16)p0;
      pl[(r + 8 * g) * 32 + 16 + c] = (__bf16)p1;
      float ps = p0 + p1;
#pragma unroll
      for (int off = 1; off < 16; off <<= 1) ps += __shfl_xor(ps, off, 32);
      lrow[r] = lrow[r] * alpha + ps;
      mrow[r] = mnew;
#pragma unroll
      for (int t = 0; t < 4; ++t) o[t][r] *= alpha;
    }
    // P (16x32 bf16) from LDS as A-fragment; V^T rows give contiguous B-fragments.
    v16bf pf = load_a(pl, 32, lane);
#pragma unroll
    for (int t = 0; t < 4; ++t) {
      v16bf vf = load_b(Vbase + (size_t)(t * 16) * L_ + k0, L_, lane);
      o[t] = WMMA_BF16(pf, vf, o[t]);
    }
  }

  __bf16* dst = Aob + ((size_t)b * L_ + qpos0) * (H_ * 64) + (size_t)h * 64;
#pragma unroll
  for (int r = 0; r < 8; ++r) {
    float inv = 1.0f / lrow[r];
#pragma unroll
    for (int t = 0; t < 4; ++t)
      dst[(size_t)(r + 8 * g) * (H_ * 64) + t * 16 + c] = (__bf16)(o[t][r] * inv);
  }
}

// ---------------- kernel 3: output projection ----------------
// Same 32Mx64N-per-wave structure with async LDS A-pipeline; fp32 output.
__global__ void __launch_bounds__(128) oproj_kernel(
    const __bf16* __restrict__ Aob, const __bf16* __restrict__ Wob,
    float* __restrict__ out) {
  const int tid  = threadIdx.x;
  const int lane = tid & 31;
  const int wave = tid >> 5;
  const int c = lane & 15, g = lane >> 4;
  const int rowTile = blockIdx.x;            // 0 .. B*L/32-1
  const int strip   = blockIdx.y * 4 + wave; // 0 .. 31

  __shared__ alignas(16) __bf16 abuf[2][32 * 64];

  const __bf16* Arow = Aob + (size_t)rowTile * 32 * HID_;
  const __bf16* Brow = Wob + (size_t)strip * 64 * HID_;

  v8f acc[2][4] = {};
  constexpr int NCH = HID_ / 64;

  stage_a_chunk(Arow, HID_, &abuf[0][0], tid);
  for (int j = 0; j < NCH; ++j) {
    if (j + 1 < NCH) {
      stage_a_chunk(Arow + (j + 1) * 64, HID_, &abuf[(j + 1) & 1][0], tid);
      WAIT_ASYNC(2);
    } else {
      WAIT_ASYNC(0);
    }
    __syncthreads();
    const __bf16* lb = &abuf[j & 1][0];
#pragma unroll
    for (int s = 0; s < 64; s += 32) {
      v16bf a0 = load_a(lb + s, 64, lane);
      v16bf a1 = load_a(lb + 16 * 64 + s, 64, lane);
#pragma unroll
      for (int t = 0; t < 4; ++t) {
        v16bf bf = load_b(Brow + (size_t)(t * 16) * HID_ + j * 64 + s, HID_, lane);
        acc[0][t] = WMMA_BF16(a0, bf, acc[0][t]);
        acc[1][t] = WMMA_BF16(a1, bf, acc[1][t]);
      }
    }
    __syncthreads();
  }

  float* drow = out + (size_t)rowTile * 32 * HID_ + strip * 64;
#pragma unroll
  for (int m = 0; m < 2; ++m)
#pragma unroll
    for (int t = 0; t < 4; ++t)
#pragma unroll
      for (int r = 0; r < 8; ++r)
        drow[(size_t)(m * 16 + r + 8 * g) * HID_ + t * 16 + c] = acc[m][t][r];
}

// ---------------- host side ----------------
extern "C" void kernel_launch(void* const* d_in, const int* in_sizes, int n_in,
                              void* d_out, int out_size, void* d_ws, size_t ws_size,
                              hipStream_t stream) {
  (void)in_sizes; (void)n_in; (void)out_size; (void)ws_size;
  const float* x    = (const float*)d_in[0];
  const float* cosb = (const float*)d_in[1];
  const float* sinb = (const float*)d_in[2];
  const float* Wq   = (const float*)d_in[3];
  const float* Wk   = (const float*)d_in[4];
  const float* Wv   = (const float*)d_in[5];
  const float* Wo   = (const float*)d_in[6];
  float* out = (float*)d_out;

  // workspace layout (bf16 elements)
  const size_t nX  = (size_t)B_ * L_ * HID_;       // 8,388,608
  const size_t nWq = (size_t)H_ * D_ * HID_;       // 4,194,304
  const size_t nWk = (size_t)HKV_ * D_ * HID_;     // 1,048,576
  const size_t nQ  = (size_t)B_ * H_ * L_ * D_;    // 8,388,608
  const size_t nK  = (size_t)B_ * HKV_ * L_ * D_;  // 2,097,152

  __bf16* ws  = (__bf16*)d_ws;
  __bf16* xb  = ws;                  size_t off = nX;
  __bf16* Wqb = ws + off;            off += nWq;
  __bf16* Wkb = ws + off;            off += nWk;
  __bf16* Wvb = ws + off;            off += nWk;
  __bf16* Wob = ws + off;            off += nWq;
  __bf16* Qb  = ws + off;            off += nQ;
  __bf16* Kb  = ws + off;            off += nK;
  __bf16* Vtb = ws + off;            off += nK;
  __bf16* Aob = ws + off;            off += nX;

  auto cvt = [&](const float* s, __bf16* d, size_t n) {
    int n4 = (int)(n / 4);
    cvt4_kernel<<<(n4 + 255) / 256, 256, 0, stream>>>(s, d, n4);
  };
  cvt(x,  xb,  nX);
  cvt(Wq, Wqb, nWq);
  cvt(Wk, Wkb, nWk);
  cvt(Wv, Wvb, nWk);
  cvt(Wo, Wob, nWq);

  // QKV + RoPE: 128 row tiles (32 rows each) x 48 strips (4 waves/block)
  qkv_rope_kernel<<<dim3(B_ * L_ / 32, 12), 128, 0, stream>>>(
      xb, Wqb, Wkb, Wvb, cosb, sinb, Qb, Kb, Vtb);

  // attention: B*H*(L/16) = 8192 waves, 4 per block
  attn_kernel<<<(B_ * H_ * (L_ / 16)) / 4, 128, 0, stream>>>(Qb, Kb, Vtb, Aob);

  // output projection: 128 row tiles x 32 strips
  oproj_kernel<<<dim3(B_ * L_ / 32, 8), 128, 0, stream>>>(Aob, Wob, out);
}
